// adj_embedding_90417651516418
// MI455X (gfx1250) — matrix operand order repaired and checked by reading, not compiled, not run on hardware
//
#include <hip/hip_runtime.h>
#include <hip/hip_bf16.h>
#include <cstdint>

typedef __bf16 bf16;
typedef __attribute__((ext_vector_type(16))) __bf16 v16bf;
typedef __attribute__((ext_vector_type(8)))  __bf16 v8bf;
typedef __attribute__((ext_vector_type(8)))  float   v8f;

#define NNODES 512
#define DIM    64
#define CONV_CH 32
#define BATCH  4
#define JDIM   (BATCH * CONV_CH * DIM)   // 8192 columns of the propagation GEMM
#define PROPA  0.05f
#define SELU_S 1.0507009873554804934193349852946f
#define SELU_A 1.6732632423543772848170429916717f

__device__ __forceinline__ float selu_f(float x) {
    return x > 0.f ? SELU_S * x : SELU_S * SELU_A * (__expf(x) - 1.f);
}

// ---------------- graph construction ----------------

// nv = tanh(3*(emb @ W^T + b)); 1024 blocks (0-511 nv1, 512-1023 nv2) x 64 thr
__global__ void k_nv(const float* __restrict__ emb1, const float* __restrict__ emb2,
                     const float* __restrict__ w1, const float* __restrict__ b1,
                     const float* __restrict__ w2, const float* __restrict__ b2,
                     float* __restrict__ nv1, float* __restrict__ nv2) {
    int blk = blockIdx.x;
    int which = blk >> 9;
    int row = blk & 511;
    int j = threadIdx.x;
    __shared__ float e[64];
    const float* emb = which ? emb2 : emb1;
    const float* w   = which ? w2   : w1;
    const float* b   = which ? b2   : b1;
    e[j] = emb[row * 64 + j];
    __syncthreads();
    float acc = b[j];
#pragma unroll 8
    for (int k = 0; k < 64; ++k) acc += e[k] * w[j * 64 + k];
    float* out = which ? nv2 : nv1;
    out[row * 64 + j] = tanhf(3.0f * acc);
}

// adj = selu(tanh(3*(nv1@nv2^T - nv2@nv1^T))); 512 blocks x 128 thr
__global__ void k_adj(const float* __restrict__ nv1, const float* __restrict__ nv2,
                      float* __restrict__ adj) {
    int v = blockIdx.x, t = threadIdx.x;
    __shared__ float s1[64], s2[64];
    if (t < 64) { s1[t] = nv1[v * 64 + t]; s2[t] = nv2[v * 64 + t]; }
    __syncthreads();
    for (int w = t; w < NNODES; w += 128) {
        float d1 = 0.f, d2 = 0.f;
#pragma unroll 8
        for (int k = 0; k < 64; ++k) {
            d1 += s1[k] * nv2[w * 64 + k];
            d2 += s2[k] * nv1[w * 64 + k];
        }
        float a = tanhf(3.0f * (d1 - d2));
        adj[v * NNODES + w] = selu_f(a);
    }
}

// per-row top-K=256 via bitonic sort threshold; 512 blocks x 256 thr
__global__ void k_topk(const float* __restrict__ adj, float* __restrict__ adjM) {
    __shared__ float s[NNODES];
    int v = blockIdx.x, t = threadIdx.x;
    s[t]       = adj[v * NNODES + t];
    s[t + 256] = adj[v * NNODES + t + 256];
    __syncthreads();
    for (int k = 2; k <= NNODES; k <<= 1) {
        for (int j = k >> 1; j > 0; j >>= 1) {
            for (int i = t; i < NNODES; i += 256) {
                int ixj = i ^ j;
                if (ixj > i) {
                    bool up = ((i & k) == 0);          // ascending overall
                    float a = s[i], b = s[ixj];
                    if (up ? (a > b) : (a < b)) { s[i] = b; s[ixj] = a; }
                }
            }
            __syncthreads();
        }
    }
    float thr = s[NNODES - NNODES / 2];                // 256th largest
    for (int w = t; w < NNODES; w += 256) {
        float x = adj[v * NNODES + w];
        adjM[v * NNODES + w] = (x >= thr) ? x : 0.f;
    }
}

// rsf[v] = 1 + row-sum, rsb[v] = 1 + col-sum; 512 blocks x 256 thr
__global__ void k_rowsum(const float* __restrict__ adjM,
                         float* __restrict__ rsf, float* __restrict__ rsb) {
    __shared__ float sr[256], sc[256];
    int v = blockIdx.x, t = threadIdx.x;
    sr[t] = adjM[v * NNODES + t] + adjM[v * NNODES + t + 256];
    sc[t] = adjM[t * NNODES + v] + adjM[(t + 256) * NNODES + v];
    __syncthreads();
    for (int o = 128; o > 0; o >>= 1) {
        if (t < o) { sr[t] += sr[t + o]; sc[t] += sc[t + o]; }
        __syncthreads();
    }
    if (t == 0) { rsf[v] = sr[0] + 1.f; rsb[v] = sc[0] + 1.f; }
}

// GEMM A-operands (transposed, row-normalized, bf16): Afwd[m][k] = a_fwd[k][m]
__global__ void k_buildA(const float* __restrict__ adjM,
                         const float* __restrict__ rsf, const float* __restrict__ rsb,
                         bf16* __restrict__ Afwd, bf16* __restrict__ Abwd) {
    int m = blockIdx.x;
    for (int k = threadIdx.x; k < NNODES; k += 256) {
        float d = (k == m) ? 1.f : 0.f;
        Afwd[m * NNODES + k] = (bf16)((adjM[k * NNODES + m] + d) / rsf[k]);
        Abwd[m * NNODES + k] = (bf16)((adjM[m * NNODES + k] + d) / rsb[k]);
    }
}

// ---------------- start conv: x[n,0,v,l] -> H[v][n,c,l] + bf16 transpose ----

__global__ void k_start(const float* __restrict__ x, const float* __restrict__ sw,
                        const float* __restrict__ sb,
                        float* __restrict__ H, bf16* __restrict__ HT) {
    int idx = blockIdx.x * 256 + threadIdx.x;     // v*8192 + j
    int v = idx >> 13, j = idx & 8191;
    int n = j >> 11, c = (j >> 6) & 31, l = j & 63;
    float val = x[(n * NNODES + v) * 64 + l] * sw[c] + sb[c];
    H[idx] = val;
    HT[(size_t)j * NNODES + v] = (bf16)val;
}

// ---------------- WMMA propagation GEMM --------------------------------------
// C[m,j] = sum_k A[m,k] * Bt[j,k]   (both operands K-contiguous)
// 32x32 output tile per wave: 2 A frags + 2 B frags -> 4 WMMAs per K-step
// (2 VMEM b128 per WMMA instead of 4 -> load-issue no longer the bottleneck).
// epilogue: Hout = alpha*X + (1-alpha)*C ; HoutT (bf16, [j][m]) for next hop.

__device__ __forceinline__ v16bf load_afrag(const bf16* __restrict__ p) {
    v8bf a0 = *(const v8bf*)p;
    v8bf a1 = *(const v8bf*)(p + 8);
    v16bf a;
#pragma unroll
    for (int i = 0; i < 8; ++i) { a[i] = a0[i]; a[i + 8] = a1[i]; }
    return a;
}

__device__ __forceinline__ void epilogue_tile(v8f c, int mbase, int moff, int j,
                                              const float* __restrict__ X,
                                              float* __restrict__ Hout,
                                              bf16* __restrict__ HoutT) {
    v8bf tb;
#pragma unroll
    for (int r = 0; r < 8; ++r) {
        int m = mbase + moff + r;
        float hv = PROPA * X[(size_t)m * JDIM + j] + (1.f - PROPA) * c[r];
        Hout[(size_t)m * JDIM + j] = hv;
        tb[r] = (bf16)hv;
    }
    if (HoutT) *(v8bf*)(HoutT + (size_t)j * NNODES + mbase + moff) = tb;
}

__global__ __launch_bounds__(256)
void k_gemm_prop(const bf16* __restrict__ A, const bf16* __restrict__ Bt,
                 const float* __restrict__ X, float* __restrict__ Hout,
                 bf16* __restrict__ HoutT) {
    const int lane = threadIdx.x & 31;
    const int wave = threadIdx.x >> 5;
    // 8 waves arranged 2(m) x 4(j): block tile = 64m x 128j
    const int m0 = blockIdx.y * 64 + (wave >> 2) * 32;
    const int j0 = blockIdx.x * 128 + (wave & 3) * 32;

    const int lm  = lane & 15;
    const int kbA = (lane < 16) ? 0 : 16;   // A frag: 16 contiguous K values
    const int kbB = (lane < 16) ? 0 : 16;   // B frag: 16 contiguous K values

    const bf16* aptr0 = A + (m0 + lm) * NNODES + kbA;
    const bf16* aptr1 = aptr0 + 16 * NNODES;
    const int jl = j0 + lm;
    const bf16* bptr0 = Bt + (size_t)jl * NNODES + kbB;
    const bf16* bptr1 = bptr0 + (size_t)16 * NNODES;

    v8f c00 = {0.f,0.f,0.f,0.f,0.f,0.f,0.f,0.f};
    v8f c01 = {0.f,0.f,0.f,0.f,0.f,0.f,0.f,0.f};
    v8f c10 = {0.f,0.f,0.f,0.f,0.f,0.f,0.f,0.f};
    v8f c11 = {0.f,0.f,0.f,0.f,0.f,0.f,0.f,0.f};

    for (int k0 = 0; k0 < NNODES; k0 += 32) {
        v16bf a0 = load_afrag(aptr0 + k0);
        v16bf a1 = load_afrag(aptr1 + k0);
        v16bf b0 = *(const v16bf*)(bptr0 + k0);
        v16bf b1 = *(const v16bf*)(bptr1 + k0);
        __builtin_prefetch(bptr0 + k0 + 128, 0, 1);
        __builtin_prefetch(bptr1 + k0 + 128, 0, 1);
        c00 = __builtin_amdgcn_wmma_f32_16x16x32_bf16(false, a0, false, b0,
                                                      (short)0, c00, false, false);
        c01 = __builtin_amdgcn_wmma_f32_16x16x32_bf16(false, a0, false, b1,
                                                      (short)0, c01, false, false);
        c10 = __builtin_amdgcn_wmma_f32_16x16x32_bf16(false, a1, false, b0,
                                                      (short)0, c10, false, false);
        c11 = __builtin_amdgcn_wmma_f32_16x16x32_bf16(false, a1, false, b1,
                                                      (short)0, c11, false, false);
    }

    // D layout: VGPR r -> M = r (+8 for lanes 16-31), N = lane&15
    const int moff = (lane < 16) ? 0 : 8;
    epilogue_tile(c00, m0,      moff, j0 + lm,      X, Hout, HoutT);
    epilogue_tile(c01, m0,      moff, j0 + 16 + lm, X, Hout, HoutT);
    epilogue_tile(c10, m0 + 16, moff, j0 + lm,      X, Hout, HoutT);
    epilogue_tile(c11, m0 + 16, moff, j0 + 16 + lm, X, Hout, HoutT);
}

// ---------------- fused mixprop MLPs (fwd+bwd concat, 96->32) ---------------

__global__ void k_mlp(const float* __restrict__ Hin, const float* __restrict__ H1f,
                      const float* __restrict__ H2f, const float* __restrict__ H1b,
                      const float* __restrict__ H2b,
                      const float* __restrict__ w1, const float* __restrict__ b1,
                      const float* __restrict__ w2, const float* __restrict__ b2,
                      float* __restrict__ Hout, bf16* __restrict__ HoutT) {
    int idx = blockIdx.x * 256 + threadIdx.x;     // v*8192 + n*2048 + o*64 + l
    int v = idx >> 13, j = idx & 8191;
    int n = j >> 11, o = (j >> 6) & 31, l = j & 63;
    size_t base = (size_t)v * JDIM + n * 2048 + l;
    const float* w1o = w1 + o * 96;
    const float* w2o = w2 + o * 96;
    float acc = b1[o] + b2[o];
#pragma unroll 4
    for (int c = 0; c < 32; ++c) {
        size_t off = base + c * 64;
        acc += (w1o[c] + w2o[c]) * Hin[off];
        acc += w1o[32 + c] * H1f[off] + w1o[64 + c] * H2f[off];
        acc += w2o[32 + c] * H1b[off] + w2o[64 + c] * H2b[off];
    }
    Hout[idx] = acc;
    HoutT[(size_t)j * NNODES + v] = (bf16)acc;
}

// ---------------- LayerNorm over L=64 (in-place), ends, output --------------

__global__ void k_ln(float* __restrict__ X, const float* __restrict__ w,
                     const float* __restrict__ b, int rows) {
    int r = blockIdx.x * 256 + threadIdx.x;
    if (r >= rows) return;
    float* x = X + (size_t)r * 64;
    float u = 0.f;
    for (int l = 0; l < 64; ++l) u += x[l];
    u *= (1.f / 64.f);
    float s = 0.f;
    for (int l = 0; l < 64; ++l) { float d = x[l] - u; s += d * d; }
    s *= (1.f / 64.f);
    float inv = rsqrtf(s + 1e-12f);
    for (int l = 0; l < 64; ++l) x[l] = w[l] * (x[l] - u) * inv + b[l];
}

__global__ void k_end1(const float* __restrict__ H, const float* __restrict__ w,
                       const float* __restrict__ b, float* __restrict__ E1) {
    int idx = blockIdx.x * 256 + threadIdx.x;     // v*16384 + n*4096 + o*64 + l
    int v = idx >> 14, rem = idx & 16383;
    int n = rem >> 12, o = (rem >> 6) & 63, l = rem & 63;
    size_t base = (size_t)v * JDIM + n * 2048 + l;
    float acc = b[o];
#pragma unroll 4
    for (int c = 0; c < 32; ++c) acc += w[o * 32 + c] * H[base + c * 64];
    E1[idx] = selu_f(acc);
}

__global__ void k_end2(const float* __restrict__ E1, const float* __restrict__ w,
                       const float* __restrict__ b, float* __restrict__ E2) {
    int idx = blockIdx.x * 256 + threadIdx.x;
    int v = idx >> 14, rem = idx & 16383;
    int n = rem >> 12, o2 = (rem >> 6) & 63, l = rem & 63;
    size_t base = (size_t)v * 16384 + n * 4096 + l;
    float acc = b[o2];
#pragma unroll 4
    for (int o = 0; o < 64; ++o) acc += w[o2 * 64 + o] * E1[base + o * 64];
    E2[idx] = acc;
}

__global__ void k_out(const float* __restrict__ E2, float* __restrict__ out) {
    int idx = blockIdx.x * 256 + threadIdx.x;     // (n*64+l)*512 + v
    int v = idx & 511, nl = idx >> 9;
    int l = nl & 63, n = nl >> 6;
    size_t base = (size_t)v * 16384 + n * 4096 + l;
    float acc = 0.f;
#pragma unroll 8
    for (int c = 0; c < 64; ++c) acc += E2[base + c * 64];
    out[idx] = acc;
}

// ---------------- host orchestration ----------------------------------------

extern "C" void kernel_launch(void* const* d_in, const int* in_sizes, int n_in,
                              void* d_out, int out_size, void* d_ws, size_t ws_size,
                              hipStream_t stream) {
    const float* x       = (const float*)d_in[0];
    const float* emb1    = (const float*)d_in[1];
    const float* emb2    = (const float*)d_in[2];
    const float* lin1_w  = (const float*)d_in[3];
    const float* lin1_b  = (const float*)d_in[4];
    const float* lin2_w  = (const float*)d_in[5];
    const float* lin2_b  = (const float*)d_in[6];
    const float* start_w = (const float*)d_in[7];
    const float* start_b = (const float*)d_in[8];
    const float* g1_w    = (const float*)d_in[9];
    const float* g1_b    = (const float*)d_in[10];
    const float* g2_w    = (const float*)d_in[11];
    const float* g2_b    = (const float*)d_in[12];
    const float* ln_w    = (const float*)d_in[13];
    const float* ln_b    = (const float*)d_in[14];
    const float* end1_w  = (const float*)d_in[15];
    const float* end1_b  = (const float*)d_in[16];
    const float* end2_w  = (const float*)d_in[17];
    const float* end2_b  = (const float*)d_in[18];
    float* out = (float*)d_out;

    // ---- carve workspace ----
    char* p = (char*)d_ws;
    auto alloc = [&](size_t bytes) -> void* {
        void* r = (void*)p;
        p += (bytes + 255) & ~(size_t)255;
        return r;
    };
    const size_t HSZ = (size_t)NNODES * JDIM;          // 4.19M elems
    float* nv1  = (float*)alloc(NNODES * DIM * 4);
    float* nv2  = (float*)alloc(NNODES * DIM * 4);
    float* adj  = (float*)alloc((size_t)NNODES * NNODES * 4);
    float* adjM = (float*)alloc((size_t)NNODES * NNODES * 4);
    float* rsf  = (float*)alloc(NNODES * 4);
    float* rsb  = (float*)alloc(NNODES * 4);
    bf16*  Afwd = (bf16*)alloc((size_t)NNODES * NNODES * 2);
    bf16*  Abwd = (bf16*)alloc((size_t)NNODES * NNODES * 2);
    float* HA   = (float*)alloc(HSZ * 4);
    float* HB   = (float*)alloc(HSZ * 4);
    float* H1f  = (float*)alloc(HSZ * 4);
    float* H2f  = (float*)alloc(HSZ * 4);              // contiguous after H1f
    float* H1b  = (float*)alloc(HSZ * 4);
    float* H2b  = (float*)alloc(HSZ * 4);              // contiguous after H1b
    bf16*  TA   = (bf16*)alloc(HSZ * 2);
    bf16*  TB   = (bf16*)alloc(HSZ * 2);
    bf16*  T1f  = (bf16*)alloc(HSZ * 2);
    bf16*  T1b  = (bf16*)alloc(HSZ * 2);
    (void)H2f; (void)H2b; (void)ws_size; (void)n_in; (void)in_sizes; (void)out_size;

    // ---- graph construction ----
    k_nv<<<1024, 64, 0, stream>>>(emb1, emb2, lin1_w, lin1_b, lin2_w, lin2_b, nv1, nv2);
    k_adj<<<NNODES, 128, 0, stream>>>(nv1, nv2, adj);
    k_topk<<<NNODES, 256, 0, stream>>>(adj, adjM);
    k_rowsum<<<NNODES, 256, 0, stream>>>(adjM, rsf, rsb);
    k_buildA<<<NNODES, 256, 0, stream>>>(adjM, rsf, rsb, Afwd, Abwd);

    // ---- start conv ----
    k_start<<<(int)(HSZ / 256), 256, 0, stream>>>(x, start_w, start_b, HA, TA);

    // ---- 2 layers of dual-direction mixprop ----
    dim3 ggrid(JDIM / 128, NNODES / 64);
    float* Hin = HA; float* Hnx = HB;
    bf16*  Tin = TA; bf16*  Tnx = TB;
    for (int layer = 0; layer < 2; ++layer) {
        k_gemm_prop<<<ggrid, 256, 0, stream>>>(Afwd, Tin, Hin, H1f, T1f);
        k_gemm_prop<<<ggrid, 256, 0, stream>>>(Afwd, T1f, Hin, H2f, (bf16*)nullptr);
        k_gemm_prop<<<ggrid, 256, 0, stream>>>(Abwd, Tin, Hin, H1b, T1b);
        k_gemm_prop<<<ggrid, 256, 0, stream>>>(Abwd, T1b, Hin, H2b, (bf16*)nullptr);
        k_mlp<<<(int)(HSZ / 256), 256, 0, stream>>>(
            Hin, H1f, H2f, H1b, H2b,
            g1_w + layer * CONV_CH * 96, g1_b + layer * CONV_CH,
            g2_w + layer * CONV_CH * 96, g2_b + layer * CONV_CH,
            Hnx, Tnx);
        { float* t = Hin; Hin = Hnx; Hnx = t; }
        { bf16* t = Tin; Tin = Tnx; Tnx = t; }
    }
    // Hin now holds final h [v][n,c(32),l]

    // ---- LN -> selu(end1) -> end2 -> LN -> channel-sum output ----
    int rows1 = NNODES * BATCH * CONV_CH;              // 65536
    k_ln<<<rows1 / 256, 256, 0, stream>>>(Hin, ln_w, ln_b, rows1);

    float* E1 = H1f;                                   // aliases H1f+H2f (2 slots)
    float* E2 = H1b;                                   // aliases H1b+H2b
    size_t esz = (size_t)NNODES * BATCH * 64 * 64;     // 8.39M
    k_end1<<<(int)(esz / 256), 256, 0, stream>>>(Hin, end1_w, end1_b, E1);
    k_end2<<<(int)(esz / 256), 256, 0, stream>>>(E1, end2_w, end2_b, E2);

    int rows2 = NNODES * BATCH * 64;                   // 131072
    k_ln<<<rows2 / 256, 256, 0, stream>>>(E2, ln_w, ln_b, rows2);

    k_out<<<(BATCH * 64 * NNODES) / 256, 256, 0, stream>>>(E2, out);
}
